// TokenMixing_9156870275238
// MI455X (gfx1250) — matrix operand order repaired and checked by reading, not compiled, not run on hardware
//
#include <hip/hip_runtime.h>
#include <hip/hip_bf16.h>
#include <cstdint>
#include <cstddef>

// Problem constants (from reference)
#define CD    1024            // D_MODEL
#define BB    8
#define TT    2048
#define MT    (BB * TT)       // 16384 rows for the GEMMs
#define NK    (CD / 32)       // k-steps per GEMM

typedef __attribute__((ext_vector_type(16))) __bf16 v16bf;
typedef __attribute__((ext_vector_type(8)))  float  v8f;
typedef __attribute__((ext_vector_type(4)))  unsigned int v4u;
typedef __attribute__((ext_vector_type(4)))  int    v4i;
typedef __attribute__((ext_vector_type(8)))  int    v8i;

#if __has_builtin(__builtin_amdgcn_tensor_load_to_lds) && \
    __has_builtin(__builtin_amdgcn_s_wait_tensorcnt)
#define USE_TDM 1
#else
#define USE_TDM 0
#endif

// ---------------------------------------------------------------------------
// float -> bf16 with round-to-nearest-even
__device__ __forceinline__ unsigned short f2bf(float f) {
    union { float f; unsigned u; } c; c.f = f;
    unsigned u = c.u;
    unsigned r = u + 0x7FFFu + ((u >> 16) & 1u);
    return (unsigned short)(r >> 16);
}

// ---------------------------------------------------------------------------
// Kernel 1: token shift + three time-mixes, emit bf16 rows [MT, CD]
__global__ __launch_bounds__(256) void mix_kernel(
    const float* __restrict__ x,
    const float* __restrict__ tmk,
    const float* __restrict__ tmv,
    const float* __restrict__ tmr,
    unsigned short* __restrict__ xk,
    unsigned short* __restrict__ xv,
    unsigned short* __restrict__ xr)
{
    int idx = blockIdx.x * blockDim.x + threadIdx.x;
    if (idx >= MT * CD) return;
    int c = idx & (CD - 1);
    int t = (idx / CD) & (TT - 1);
    float xc = x[idx];
    float xs = (t == 0) ? 0.0f : x[idx - CD];   // token shift (zero-pad at t=0)
    float mk = tmk[c], mv = tmv[c], mr = tmr[c];
    xk[idx] = f2bf(xc * mk + xs * (1.0f - mk));
    xv[idx] = f2bf(xc * mv + xs * (1.0f - mv));
    xr[idx] = f2bf(xc * mr + xs * (1.0f - mr));
}

// ---------------------------------------------------------------------------
// Kernel 2: generic f32 -> bf16 conversion (weight matrices)
__global__ __launch_bounds__(256) void cvt_bf16_kernel(
    const float* __restrict__ in, unsigned short* __restrict__ out, int n)
{
    int idx = blockIdx.x * blockDim.x + threadIdx.x;
    if (idx < n) out[idx] = f2bf(in[idx]);
}

// ---------------------------------------------------------------------------
// TDM: 2D tile load (rows x tile_k bf16 elements, row stride row_stride
// elements) from global into LDS.  D# per CDNA5 ISA §8; row-major tile walk
// => contiguous LDS rows of tile_k elements.
__device__ __forceinline__ void tdm_load_2d(unsigned lds_addr, const void* gptr,
                                            int rows, int tile_k, int row_stride)
{
#if USE_TDM
    unsigned long long ga = (unsigned long long)(uintptr_t)gptr;
    v4u g0;
    g0.x = 1u;                                            // count=1, user D#
    g0.y = lds_addr;                                      // LDS byte address
    g0.z = (unsigned)ga;                                  // global_addr[31:0]
    g0.w = ((unsigned)(ga >> 32) & 0x01FFFFFFu)           // global_addr[56:32]
         | (2u << 30);                                    // type = 2 ("image")
    v8i g1;
    g1[0] = 1 << 16;                                      // data_size = 2 bytes
    g1[1] = (row_stride & 0xFFFF) << 16;                  // tensor_dim0[15:0]
    g1[2] = ((row_stride >> 16) & 0xFFFF)                 // tensor_dim0[31:16]
          | ((rows & 0xFFFF) << 16);                      // tensor_dim1[15:0]
    g1[3] = (tile_k & 0xFFFF) << 16;                      // tile_dim0
    g1[4] = rows & 0xFFFF;                                // tile_dim1 (tile_dim2=0)
    g1[5] = row_stride;                                   // tensor_dim0_stride[31:0]
    g1[6] = 0;                                            // stride hi / dim1_stride lo
    g1[7] = 0;                                            // dim1_stride hi
    v4i z4 = {0, 0, 0, 0};
#if __clang_major__ >= 23
    v8i z8 = {0, 0, 0, 0, 0, 0, 0, 0};
    __builtin_amdgcn_tensor_load_to_lds(g0, g1, z4, z4, z8, 0);
#else
    __builtin_amdgcn_tensor_load_to_lds(g0, g1, z4, z4, 0);
#endif
#else
    (void)lds_addr; (void)gptr; (void)rows; (void)tile_k; (void)row_stride;
#endif
}

// ---------------------------------------------------------------------------
// Kernel 3/5: WMMA GEMM  D[m,n] = sum_c A[m,c] * W[n,c]
//   A: [MT, CD] bf16 row-major,  W: [CD, CD] bf16 row-major (row = out dim n)
//   Block = 256 thr = 8 waves; block tile 256(M) x 64(N); wave tile 32 x 64
//   (8 WMMAs per k-step: 2 A-frags x 4 B-frags).  K-step 32.
//   A(256x32) and B(64x32) tiles double-buffered in LDS, moved by the TDM
//   (wave0 issues A, wave1 issues B; every wave waits tensorcnt -> no-op for
//   non-issuers since TENSORcnt is per-wave).  Fully unrolled k-loop so the
//   accumulators carry no loop phis (kills the v_mov/v_nop storm).
//   Fragment reads from LDS per ISA 7.12.2 layouts.
__global__ __launch_bounds__(256) void wmma_gemm_kernel(
    const unsigned short* __restrict__ A,
    const unsigned short* __restrict__ W,
    float* __restrict__ D)
{
    __shared__ __align__(16) unsigned short ldsA[2][256 * 32];  // 2 x 16 KB
    __shared__ __align__(16) unsigned short ldsB[2][64 * 32];   // 2 x  4 KB

    const int lane  = threadIdx.x & 31;
    const int wave  = threadIdx.x >> 5;
    const int g     = lane >> 4;
    const int l16   = lane & 15;
    const int m_blk = blockIdx.x * 256;
    const int n0    = blockIdx.y * 64;

    auto stage = [&](int buf, int k0) {
#if USE_TDM
        if (wave == 0)                                   // wave-uniform branch
            tdm_load_2d((unsigned)(uintptr_t)&ldsA[buf][0],
                        A + (size_t)m_blk * CD + k0, 256, 32, CD);
        if (wave == 1)
            tdm_load_2d((unsigned)(uintptr_t)&ldsB[buf][0],
                        W + (size_t)n0 * CD + k0, 64, 32, CD);
#else
        // Fallback: 256 threads copy 16B chunks (A: 1024 chunks, B: 256)
        const int t = threadIdx.x;
#pragma unroll
        for (int i = 0; i < 4; ++i) {
            int ch = t + i * 256;
            int row = ch >> 2, c4 = ch & 3;
            *(uint4*)&ldsA[buf][row * 32 + c4 * 8] =
                *(const uint4*)(A + (size_t)(m_blk + row) * CD + k0 + c4 * 8);
        }
        int row = t >> 2, c4 = t & 3;
        *(uint4*)&ldsB[buf][row * 32 + c4 * 8] =
            *(const uint4*)(W + (size_t)(n0 + row) * CD + k0 + c4 * 8);
#endif
    };

    v8f acc[8] = {};   // acc[mi*4+j]: mi = M sub-tile (0,1), j = N sub-tile

    stage(0, 0);

#pragma unroll
    for (int kidx = 0; kidx < NK; ++kidx) {
        const int buf = kidx & 1;
#if USE_TDM
        __builtin_amdgcn_s_wait_tensorcnt(0);   // per-wave counter: no-op for
#endif                                          // waves that issued nothing
        __syncthreads();                        // tiles for `buf` now visible

        if (kidx + 1 < NK) stage(buf ^ 1, (kidx + 1) * 32);  // overlap DMA

        union { uint4 u[2]; v16bf v; } af[2];
#pragma unroll
        for (int mi = 0; mi < 2; ++mi) {
            const unsigned short* ar =
                &ldsA[buf][(wave * 32 + mi * 16 + l16) * 32];
            af[mi].u[0] = *(const uint4*)(ar + 8 * g);
            af[mi].u[1] = *(const uint4*)(ar + 16 + 8 * g);
        }
#pragma unroll
        for (int j = 0; j < 4; ++j) {
            union { uint4 u[2]; v16bf v; } bfr;
            const unsigned short* br = &ldsB[buf][(j * 16 + l16) * 32 + 8 * g];
            bfr.u[0] = *(const uint4*)(br);
            bfr.u[1] = *(const uint4*)(br + 16);
#pragma unroll
            for (int mi = 0; mi < 2; ++mi)
                acc[mi * 4 + j] = __builtin_amdgcn_wmma_f32_16x16x32_bf16(
                    /*neg_a=*/false, af[mi].v, /*neg_b=*/false, bfr.v,
                    /*c_mod=*/(short)0, acc[mi * 4 + j],
                    /*reuse_a=*/false, /*reuse_b=*/false);
        }
    }

#pragma unroll
    for (int mi = 0; mi < 2; ++mi) {
        const int m_base = m_blk + wave * 32 + mi * 16;
#pragma unroll
        for (int j = 0; j < 4; ++j) {
            const int n = n0 + j * 16 + l16;
#pragma unroll
            for (int v = 0; v < 8; ++v) {
                D[(size_t)(m_base + v + 8 * g) * CD + n] = acc[mi * 4 + j][v];
            }
        }
    }
}

// ---------------------------------------------------------------------------
// Kernel 4: WKV recurrence (max-trick), fused sigmoid(R)*wkv -> bf16, hidden.
// One lane per (b, c) channel; wave covers 32 consecutive c -> coalesced rows.
__global__ __launch_bounds__(256) void wkv_kernel(
    const float* __restrict__ Wd,     // decay_speed [C]
    const float* __restrict__ Ud,     // U [C]
    const float* __restrict__ Kx,     // [B,T,C] f32
    const float* __restrict__ Vx,     // [B,T,C] f32
    const float* __restrict__ Rx,     // [B,T,C] f32
    unsigned short* __restrict__ rwkv_bf,  // [B,T,C] bf16 out
    float* __restrict__ hidden)            // [B,C] f32 out
{
    int idx = blockIdx.x * blockDim.x + threadIdx.x;   // over B*C
    if (idx >= BB * CD) return;
    const int c = idx & (CD - 1);
    const int b = idx >> 10;   // CD == 1024

    const float w = -__expf(Wd[c]);
    const float u = Ud[c];
    float aa = 0.0f, bb = 0.0f, pp = -1e38f;

    const size_t base = (size_t)b * TT * CD + c;
    for (int t = 0; t < TT; ++t) {
        const size_t o = base + (size_t)t * CD;
        const float kt = Kx[o];
        const float vt = Vx[o];

        const float ww = u + kt;
        const float p  = fmaxf(pp, ww);
        const float e1 = __expf(pp - p);
        const float e2 = __expf(ww - p);
        const float y  = (e1 * aa + e2 * vt) / (e1 * bb + e2);

        const float r  = Rx[o];
        const float sr = 1.0f / (1.0f + __expf(-r));
        rwkv_bf[o] = f2bf(sr * y);
        if (t == TT - 1) hidden[b * CD + c] = y;

        const float ww2 = pp + w;
        const float p2  = fmaxf(ww2, kt);
        const float e1b = __expf(ww2 - p2);
        const float e2b = __expf(kt - p2);
        aa = e1b * aa + e2b * vt;
        bb = e1b * bb + e2b;
        pp = p2;
    }
}

// ---------------------------------------------------------------------------
extern "C" void kernel_launch(void* const* d_in, const int* in_sizes, int n_in,
                              void* d_out, int out_size, void* d_ws, size_t ws_size,
                              hipStream_t stream)
{
    (void)in_sizes; (void)n_in; (void)out_size; (void)ws_size;

    const float* x   = (const float*)d_in[0];
    const float* W   = (const float*)d_in[1];
    const float* U   = (const float*)d_in[2];
    const float* tmk = (const float*)d_in[3];
    const float* tmv = (const float*)d_in[4];
    const float* tmr = (const float*)d_in[5];
    const float* Wk  = (const float*)d_in[6];
    const float* Wv  = (const float*)d_in[7];
    const float* Wr  = (const float*)d_in[8];
    const float* Wo  = (const float*)d_in[9];

    float* out    = (float*)d_out;            // [B,T,C] f32
    float* hidden = out + (size_t)MT * CD;    // [B,C]   f32 (tuple concat)

    // Workspace carve-up (256B-aligned chunks)
    char* ws = (char*)d_ws;
    size_t off = 0;
    auto carve = [&](size_t bytes) -> void* {
        void* p = ws + off;
        off += (bytes + 255) & ~(size_t)255;
        return p;
    };
    unsigned short* xk_bf = (unsigned short*)carve((size_t)MT * CD * 2);  // 32 MB
    unsigned short* xv_bf = (unsigned short*)carve((size_t)MT * CD * 2);
    unsigned short* xr_bf = (unsigned short*)carve((size_t)MT * CD * 2);
    unsigned short* Wk_bf = (unsigned short*)carve((size_t)CD * CD * 2);  // 2 MB
    unsigned short* Wv_bf = (unsigned short*)carve((size_t)CD * CD * 2);
    unsigned short* Wr_bf = (unsigned short*)carve((size_t)CD * CD * 2);
    unsigned short* Wo_bf = (unsigned short*)carve((size_t)CD * CD * 2);
    float* Kf = (float*)carve((size_t)MT * CD * 4);                        // 64 MB
    float* Vf = (float*)carve((size_t)MT * CD * 4);
    float* Rf = (float*)carve((size_t)MT * CD * 4);
    // rwkv (bf16) reuses the xk buffer: xk is dead after the K projection GEMM.
    unsigned short* rwkv_bf = xk_bf;

    // 1) mixes -> bf16
    {
        int n = MT * CD;
        mix_kernel<<<(n + 255) / 256, 256, 0, stream>>>(x, tmk, tmv, tmr,
                                                        xk_bf, xv_bf, xr_bf);
    }
    // 2) weights -> bf16
    {
        int n = CD * CD;
        int gb = (n + 255) / 256;
        cvt_bf16_kernel<<<gb, 256, 0, stream>>>(Wk, Wk_bf, n);
        cvt_bf16_kernel<<<gb, 256, 0, stream>>>(Wv, Wv_bf, n);
        cvt_bf16_kernel<<<gb, 256, 0, stream>>>(Wr, Wr_bf, n);
        cvt_bf16_kernel<<<gb, 256, 0, stream>>>(Wo, Wo_bf, n);
    }
    // 3) projection GEMMs: K, V, R   (grid: M/256 x N/64)
    {
        dim3 grid(MT / 256, CD / 64);
        wmma_gemm_kernel<<<grid, 256, 0, stream>>>(xk_bf, Wk_bf, Kf);
        wmma_gemm_kernel<<<grid, 256, 0, stream>>>(xv_bf, Wv_bf, Vf);
        wmma_gemm_kernel<<<grid, 256, 0, stream>>>(xr_bf, Wr_bf, Rf);
    }
    // 4) WKV recurrence + sigmoid(R)*wkv -> bf16, hidden
    {
        int n = BB * CD;
        wkv_kernel<<<(n + 255) / 256, 256, 0, stream>>>(W, U, Kf, Vf, Rf,
                                                        rwkv_bf, hidden);
    }
    // 5) output projection straight into d_out
    {
        dim3 grid(MT / 256, CD / 64);
        wmma_gemm_kernel<<<grid, 256, 0, stream>>>(rwkv_bf, Wo_bf, out);
    }
}
